// TopKSparseAutoencoder_40913858462059
// MI455X (gfx1250) — matrix operand order repaired and checked by reading, not compiled, not run on hardware
//
#include <hip/hip_runtime.h>
#include <hip/hip_bf16.h>
#include <stdint.h>

// ---------------------------------------------------------------------------
// TopK Sparse Autoencoder forward for MI455X (gfx1250, wave32, WMMA)
//   encode:  pre = relu(x @ W_enc^T + b_enc)      [8192, 32768]  (bf16 WMMA,
//            async-to-LDS double-buffered K pipeline)
//   topk  :  per-row radix-select of top-64, sparsify h in place
//   decode:  recon = h_sparse @ W_dec^T + b_dec   [8192, 768]    (L2-resident gather)
// d_out = recon (8192*768 floats) || h (8192*32768 floats)
// ---------------------------------------------------------------------------

#define INPUT_DIM 768
#define SAE_DIM   32768
#define KTOP      64
#define BATCH     8192

typedef __attribute__((ext_vector_type(16))) __bf16 v16bf;
typedef __attribute__((ext_vector_type(8)))  __bf16 v8bf;
typedef __attribute__((ext_vector_type(8)))  float  v8f;
typedef __attribute__((ext_vector_type(4)))  float  v4f;
typedef __attribute__((ext_vector_type(4)))  int    v4i;

// gfx1250 async global->LDS path (ASYNCcnt-tracked), with compile-safe fallback
#if defined(__has_builtin)
#if __has_builtin(__builtin_amdgcn_global_load_async_to_lds_b128) && \
    __has_builtin(__builtin_amdgcn_s_wait_asynccnt)
#define USE_ASYNC_LDS 1
#endif
#endif
#ifndef USE_ASYNC_LDS
#define USE_ASYNC_LDS 0
#endif

#if USE_ASYNC_LDS
__device__ __forceinline__ void async_copy_b128(const __bf16* g, void* l)
{
  __builtin_amdgcn_global_load_async_to_lds_b128(
      (__attribute__((address_space(1))) v4i*)(void*)g,
      (__attribute__((address_space(3))) v4i*)l,
      0, 0);
}
#endif

// ---------------- f32 -> bf16 (RNE) conversion, 4 elems/thread --------------
__global__ __launch_bounds__(256) void f32_to_bf16_kernel(
    const float* __restrict__ src, unsigned short* __restrict__ dst, int n4)
{
  int t = blockIdx.x * blockDim.x + threadIdx.x;
  if (t >= n4) return;
  v4f v = *(const v4f*)(src + (size_t)t * 4);
  unsigned short o[4];
#pragma unroll
  for (int j = 0; j < 4; j++) {
    unsigned u = __float_as_uint(v[j]);
    unsigned r = u + 0x7FFFu + ((u >> 16) & 1u);
    o[j] = (unsigned short)(r >> 16);
  }
  *(uint64_t*)(dst + (size_t)t * 4) =
      (uint64_t)o[0] | ((uint64_t)o[1] << 16) | ((uint64_t)o[2] << 32) | ((uint64_t)o[3] << 48);
}

// ---------------- encoder GEMM: bf16 WMMA, bias + ReLU ----------------------
// Block tile 64(M) x 256(N), 8 waves arranged 2x4, each wave 32x64 (2x4 WMMA).
// K pipeline: double-buffered LDS tiles filled by async global->LDS copies.
#define BLK_M 64
#define BLK_N 256
#define LDS_ROW 80           // 64B of K data + 16B pad (20 dwords -> conflict-free)

__global__ __launch_bounds__(256) void sae_encode_wmma_kernel(
    const __bf16* __restrict__ Xb,   // [BATCH, INPUT_DIM] bf16
    const __bf16* __restrict__ Wb,   // [SAE_DIM, INPUT_DIM] bf16
    const float*  __restrict__ b_enc,
    float*        __restrict__ H)    // [BATCH, SAE_DIM] relu(pre)
{
  const int lane = threadIdx.x & 31;
  const int wave = threadIdx.x >> 5;
  const int wm   = wave >> 2;              // 0..1
  const int wn   = wave & 3;               // 0..3
  const int row0 = blockIdx.x * BLK_M + wm * 32;
  const int col0 = blockIdx.y * BLK_N + wn * 64;

  const int lm = lane & 15;                // M / N index within 16
  const int hi = lane >> 4;                // K half selector

  v8f acc[2][4];
#pragma unroll
  for (int i = 0; i < 2; i++)
#pragma unroll
    for (int j = 0; j < 4; j++) acc[i][j] = {};

#if USE_ASYNC_LDS
  // --- double-buffered LDS tiles: A 64x32 bf16, B 256x32 bf16 (rows padded) ---
  __shared__ __align__(16) char sA[2][BLK_M * LDS_ROW];   //  2 x  5 KB
  __shared__ __align__(16) char sB[2][BLK_N * LDS_ROW];   //  2 x 20 KB

  const int t    = threadIdx.x;
  const int trow = t >> 2;                 // 0..63
  const int kq   = t & 3;                  // 16B chunk within the 64B K-slab
  const __bf16* gA = Xb + (size_t)(blockIdx.x * BLK_M + trow) * INPUT_DIM + kq * 8;
  const __bf16* gB = Wb + (size_t)(blockIdx.y * BLK_N + trow) * INPUT_DIM + kq * 8;
  const int lA_off = trow * LDS_ROW + kq * 16;

  auto issue_tile = [&](int buf, int k) {
    async_copy_b128(gA + k, &sA[buf][lA_off]);
#pragma unroll
    for (int i = 0; i < 4; i++)
      async_copy_b128(gB + (size_t)i * 64 * INPUT_DIM + k,
                      &sB[buf][(trow + i * 64) * LDS_ROW + kq * 16]);
  };

  issue_tile(0, 0);

  const int KSTEPS = INPUT_DIM / 32;
  for (int kk = 0; kk < KSTEPS; kk++) {
    const int cur = kk & 1;
    __builtin_amdgcn_s_wait_asynccnt(0);   // my async copies for buf[cur] done
    __syncthreads();                       // everyone's copies landed; prev reads done
    if (kk + 1 < KSTEPS) issue_tile(cur ^ 1, (kk + 1) * 32);

    const char* A = sA[cur];
    const char* B = sB[cur];
    v16bf a[2];
#pragma unroll
    for (int i = 0; i < 2; i++) {
      union { v16bf v; v8bf h[2]; } u;
      const char* p = A + (wm * 32 + i * 16 + lm) * LDS_ROW + hi * 32;
      u.h[0] = *(const v8bf*)(p);
      u.h[1] = *(const v8bf*)(p + 16);
      a[i] = u.v;
    }
    v16bf b[4];
#pragma unroll
    for (int j = 0; j < 4; j++)
      b[j] = *(const v16bf*)(B + (wn * 64 + j * 16 + lm) * LDS_ROW + hi * 32);

#pragma unroll
    for (int i = 0; i < 2; i++)
#pragma unroll
      for (int j = 0; j < 4; j++)
        acc[i][j] = __builtin_amdgcn_wmma_f32_16x16x32_bf16(
            false, a[i], false, b[j], (short)0, acc[i][j], false, false);
  }
#else
  // --- fallback: direct global fragment loads ---
  const __bf16* xrow0 = Xb + (size_t)(row0 + lm)      * INPUT_DIM + hi * 16;
  const __bf16* xrow1 = Xb + (size_t)(row0 + 16 + lm) * INPUT_DIM + hi * 16;
  const __bf16* wc[4];
#pragma unroll
  for (int j = 0; j < 4; j++)
    wc[j] = Wb + (size_t)(col0 + j * 16 + lm) * INPUT_DIM + hi * 16;

  for (int k = 0; k < INPUT_DIM; k += 32) {
    if (k + 32 < INPUT_DIM) {
      __builtin_prefetch(xrow0 + k + 32, 0, 1);
      __builtin_prefetch(wc[0] + k + 32, 0, 1);
      __builtin_prefetch(wc[2] + k + 32, 0, 1);
    }
    v16bf a[2];
    {
      union { v16bf v; v8bf h[2]; } u;
      u.h[0] = *(const v8bf*)(xrow0 + k);
      u.h[1] = *(const v8bf*)(xrow0 + k + 8);
      a[0] = u.v;
      u.h[0] = *(const v8bf*)(xrow1 + k);
      u.h[1] = *(const v8bf*)(xrow1 + k + 8);
      a[1] = u.v;
    }
    v16bf b[4];
#pragma unroll
    for (int j = 0; j < 4; j++) b[j] = *(const v16bf*)(wc[j] + k);
#pragma unroll
    for (int i = 0; i < 2; i++)
#pragma unroll
      for (int j = 0; j < 4; j++)
        acc[i][j] = __builtin_amdgcn_wmma_f32_16x16x32_bf16(
            false, a[i], false, b[j], (short)0, acc[i][j], false, false);
  }
#endif

  // Epilogue: C/D layout — VGPR r: lanes 0-15 => M=r, lanes 16-31 => M=r+8; N = lane&15
#pragma unroll
  for (int i = 0; i < 2; i++) {
    const int mbase = row0 + i * 16 + hi * 8;
#pragma unroll
    for (int j = 0; j < 4; j++) {
      const int col  = col0 + j * 16 + lm;
      const float bias = b_enc[col];
      float* out = H + (size_t)mbase * SAE_DIM + col;
#pragma unroll
      for (int r = 0; r < 8; r++) {
        float v = acc[i][j][r] + bias;
        out[(size_t)r * SAE_DIM] = v > 0.f ? v : 0.f;
      }
    }
  }
}

// ---------------- per-row top-64 via 4x8-bit radix select -------------------
// Post-ReLU values are >= 0, so their float bit patterns are monotone as u32.
__global__ __launch_bounds__(256) void topk_kernel(
    float* __restrict__ H,
    float* __restrict__ vals, int* __restrict__ idxs, int* __restrict__ counts)
{
  const int row = blockIdx.x;
  float* hrow = H + (size_t)row * SAE_DIM;

  __shared__ unsigned hist[256];
  __shared__ unsigned s_sel, s_rem;
  __shared__ unsigned s_eq, s_cnt;

  unsigned prefix = 0, prefix_bits = 0, remaining = KTOP;

  for (int pass = 0; pass < 4; pass++) {
    const int shift = 24 - pass * 8;
    hist[threadIdx.x] = 0;
    __syncthreads();
    const unsigned mask_hi = prefix_bits ? (0xFFFFFFFFu << (32u - prefix_bits)) : 0u;
    for (int i = threadIdx.x; i < SAE_DIM; i += 256) {
      unsigned u = __float_as_uint(hrow[i]);
      if ((u & mask_hi) == prefix)
        atomicAdd(&hist[(u >> shift) & 255u], 1u);
    }
    __syncthreads();
    if (threadIdx.x == 0) {
      unsigned cum = 0, sel = 0, rem = remaining;
      for (int b = 255; b >= 0; b--) {
        unsigned c = hist[b];
        if (cum + c >= rem) { sel = (unsigned)b; rem = rem - cum; break; }
        cum += c;
      }
      s_sel = sel; s_rem = rem;
    }
    __syncthreads();
    prefix |= s_sel << shift;
    prefix_bits += 8;
    remaining = s_rem;
    __syncthreads();
  }

  if (threadIdx.x == 0) { s_eq = 0; s_cnt = 0; }
  __syncthreads();

  const unsigned T = prefix;    // bit pattern of 64th-largest value
  float* vrow = vals + (size_t)row * KTOP;
  int*   irow = idxs + (size_t)row * KTOP;
  for (int i = threadIdx.x; i < SAE_DIM; i += 256) {
    float f = hrow[i];
    unsigned u = __float_as_uint(f);
    bool keep;
    if (u > T)       keep = true;
    else if (u == T) keep = (atomicAdd(&s_eq, 1u) < remaining);
    else             keep = false;
    if (keep) {
      unsigned p = atomicAdd(&s_cnt, 1u);
      if (p < KTOP) { vrow[p] = f; irow[p] = i; }
    } else {
      hrow[i] = 0.f;
    }
  }
  __syncthreads();
  if (threadIdx.x == 0) {
    unsigned c = s_cnt;
    counts[row] = (int)(c < KTOP ? c : (unsigned)KTOP);
  }
}

// ---------------- sparse decode: recon = sum val_k * W_dec[:, idx_k] + b ----
__global__ __launch_bounds__(256) void decode_kernel(
    const float* __restrict__ vals, const int* __restrict__ idxs, const int* __restrict__ counts,
    const float* __restrict__ Wdec,   // [INPUT_DIM, SAE_DIM]
    const float* __restrict__ b_dec,
    float* __restrict__ recon)        // [BATCH, INPUT_DIM]
{
  const int row = blockIdx.x;
  __shared__ float sv[KTOP];
  __shared__ int   si[KTOP];
  const int cnt = counts[row];
  if (threadIdx.x < KTOP) {
    const bool ok = (int)threadIdx.x < cnt;
    sv[threadIdx.x] = ok ? vals[(size_t)row * KTOP + threadIdx.x] : 0.f;
    si[threadIdx.x] = ok ? idxs[(size_t)row * KTOP + threadIdx.x] : 0;
  }
  __syncthreads();
  for (int i = threadIdx.x; i < INPUT_DIM; i += 256) {
    float acc = b_dec[i];
    const float* wrow = Wdec + (size_t)i * SAE_DIM;   // W_dec resident in 192MB L2
#pragma unroll 8
    for (int k = 0; k < KTOP; k++) acc += sv[k] * wrow[si[k]];
    recon[(size_t)row * INPUT_DIM + i] = acc;
  }
}

// ---------------------------------------------------------------------------
extern "C" void kernel_launch(void* const* d_in, const int* in_sizes, int n_in,
                              void* d_out, int out_size, void* d_ws, size_t ws_size,
                              hipStream_t stream)
{
  (void)in_sizes; (void)n_in; (void)out_size; (void)ws_size;
  const float* x     = (const float*)d_in[0];
  const float* W_enc = (const float*)d_in[1];
  const float* b_enc = (const float*)d_in[2];
  const float* W_dec = (const float*)d_in[3];
  const float* b_dec = (const float*)d_in[4];

  float* recon = (float*)d_out;                              // [BATCH, INPUT_DIM]
  float* H     = (float*)d_out + (size_t)BATCH * INPUT_DIM;  // [BATCH, SAE_DIM]

  // Workspace layout (~67 MB): bf16 X, bf16 W_enc, topk vals/idxs/counts
  char* ws = (char*)d_ws;
  size_t off = 0;
  unsigned short* Xb = (unsigned short*)(ws + off); off += (size_t)BATCH * INPUT_DIM * 2;
  off = (off + 255) & ~(size_t)255;
  unsigned short* Wb = (unsigned short*)(ws + off); off += (size_t)SAE_DIM * INPUT_DIM * 2;
  off = (off + 255) & ~(size_t)255;
  float* tk_vals = (float*)(ws + off); off += (size_t)BATCH * KTOP * 4;
  int*   tk_idxs = (int*)(ws + off);   off += (size_t)BATCH * KTOP * 4;
  int*   tk_cnts = (int*)(ws + off);   off += (size_t)BATCH * 4;

  // 1) convert to bf16
  {
    int n4 = (BATCH * INPUT_DIM) / 4;
    f32_to_bf16_kernel<<<(n4 + 255) / 256, 256, 0, stream>>>(x, Xb, n4);
  }
  {
    int n4 = (SAE_DIM * INPUT_DIM) / 4;
    f32_to_bf16_kernel<<<(n4 + 255) / 256, 256, 0, stream>>>(W_enc, Wb, n4);
  }

  // 2) encoder GEMM (WMMA) -> relu(pre) into H region of d_out
  {
    dim3 grid(BATCH / BLK_M, SAE_DIM / BLK_N);
    sae_encode_wmma_kernel<<<grid, 256, 0, stream>>>(
        (const __bf16*)Xb, (const __bf16*)Wb, b_enc, H);
  }

  // 3) per-row top-64: sparsify H in place + compact (val,idx) list
  topk_kernel<<<BATCH, 256, 0, stream>>>(H, tk_vals, tk_idxs, tk_cnts);

  // 4) sparse decode into recon region
  decode_kernel<<<BATCH, 256, 0, stream>>>(tk_vals, tk_idxs, tk_cnts, W_dec, b_dec, recon);
}
